// FocalContrastiveLoss_34462817583459
// MI455X (gfx1250) — matrix operand order repaired and checked by reading, not compiled, not run on hardware
//
#include <hip/hip_runtime.h>

// ---------------------------------------------------------------------------
// Focal contrastive loss, fused, for MI455X (gfx1250, wave32, WMMA).
//   sim = (F F^T) / T  via split-bf16 WMMA (hi/lo 3-product emulation, f32 acc)
//   pass 0: per-row neg_exp = sum_{labels differ} exp(sim)
//   pass 1: per-row sum over positives of -(1-pt)^2 * log(pt+eps)
//   finalize: row_loss = row_sum / pos_cnt (valid rows), mean over B
//
// Compute-bound: 2 * 8192^2 * 512 * 2 passes = 137 GFLOP vs 16 MB of HBM
// traffic (features are L2-resident; sim is never materialized).
// 3 independent WMMA accumulator chains hide matrix-pipe latency at the
// 1-workgroup/WGP occupancy forced by the 267 KB LDS working set.
// ---------------------------------------------------------------------------

#define BB    8192          // batch
#define DD    512           // feature dim
#define ROWS  64            // rows per workgroup
#define PAN   32            // columns per panel
#define LDA   520           // padded k-stride in ushorts (512 + 8): 4-bank skew
#define NPAN  (BB / PAN)    // 256 panels
#define NKS   (DD / 32)     // 16 k-steps of 32
#define INV_T 10.0f
#define EPSV  1e-12f

typedef __attribute__((ext_vector_type(16))) __bf16 v16bf;
typedef __attribute__((ext_vector_type(8)))  float  v8f;

// ---- bf16 helpers ----------------------------------------------------------
__device__ __forceinline__ unsigned short bf16_rne(float f) {
  unsigned int u = __float_as_uint(f);
  u += 0x7FFFu + ((u >> 16) & 1u);          // round-to-nearest-even truncation
  return (unsigned short)(u >> 16);
}

__device__ __forceinline__ void split2(float f, unsigned short& h, unsigned short& l) {
  h = bf16_rne(f);
  float fh = __uint_as_float(((unsigned int)h) << 16);
  l = bf16_rne(f - fh);                     // residual: next ~8 mantissa bits
}

// Load a v16bf WMMA operand as two aligned 16B LDS reads (ds_load_b128).
__device__ __forceinline__ v16bf ldmat(const unsigned short* p0, const unsigned short* p1) {
  union { uint4 q[2]; v16bf v; } u;
  u.q[0] = *(const uint4*)p0;
  u.q[1] = *(const uint4*)p1;
  return u.v;
}

__device__ __forceinline__ v8f wmma_bf16(v16bf a, v16bf b, v8f c) {
  // (neg_a, A, neg_b, B, c_mod, C, reuse_a, reuse_b)
  return __builtin_amdgcn_wmma_f32_16x16x32_bf16(false, a, false, b, (short)0, c,
                                                 false, false);
}

// Stage one 32-column panel: global f32 -> LDS bf16 hi/lo, column-major-per-col.
__device__ __forceinline__ void stage_panel(const float* __restrict__ feats, int pan,
                                            unsigned short* bhi, unsigned short* blo,
                                            int tid) {
  const int vecs = PAN * DD / 4;            // 4096 float4 slots
  for (int i = tid; i < vecs; i += 256) {
    const int c  = i >> 7;                  // i / 128
    const int kv = i & 127;
    const float4 f =
        ((const float4*)feats)[(((size_t)(pan * PAN + c)) * DD >> 2) + kv];
    unsigned short h0, h1, h2, h3, l0, l1, l2, l3;
    split2(f.x, h0, l0); split2(f.y, h1, l1);
    split2(f.z, h2, l2); split2(f.w, h3, l3);
    uint2 ph; ph.x = (unsigned)h0 | ((unsigned)h1 << 16);
              ph.y = (unsigned)h2 | ((unsigned)h3 << 16);
    uint2 pl; pl.x = (unsigned)l0 | ((unsigned)l1 << 16);
              pl.y = (unsigned)l2 | ((unsigned)l3 << 16);
    *(uint2*)(bhi + (size_t)c * LDA + kv * 4) = ph;
    *(uint2*)(blo + (size_t)c * LDA + kv * 4) = pl;
  }
}

// ---------------------------------------------------------------------------
__global__ __launch_bounds__(256, 1)
void fcl_main(const float* __restrict__ feats,
              const long long* __restrict__ labels,
              float* __restrict__ partials) {
  extern __shared__ __align__(16) char smem[];
  unsigned short* Ahi = (unsigned short*)smem;              // ROWS*LDA
  unsigned short* Alo = Ahi + (size_t)ROWS * LDA;           // ROWS*LDA
  unsigned short* Bbuf = Alo + (size_t)ROWS * LDA;          // 2 bufs x (hi+lo) PAN*LDA
  float* rparts  = (float*)(Bbuf + (size_t)4 * PAN * LDA);  // 2*ROWS
  float* lossbuf = rparts + 2 * ROWS;                       // ROWS
  int*   hist    = (int*)(lossbuf + ROWS);                  // 16

  const int tid  = threadIdx.x;
  const int wave = tid >> 5;
  const int lane = tid & 31;
  const int half = lane >> 4;        // wave32: two 16-lane halves
  const int m    = lane & 15;
  const int blk  = blockIdx.x;
  const int row0 = blk * ROWS;

  // --- label histogram (16 classes) ---
  if (tid < 16) hist[tid] = 0;
  __syncthreads();
  for (int i = tid; i < BB; i += 256) atomicAdd(&hist[(int)labels[i]], 1);

  // --- stage A block: 64 x 512 f32 -> bf16 hi/lo in LDS (resident all kernel) ---
  {
    const int vecs = ROWS * DD / 4;
    for (int i = tid; i < vecs; i += 256) {
      const int r  = i >> 7;
      const int kv = i & 127;
      const float4 f =
          ((const float4*)feats)[(((size_t)(row0 + r)) * DD >> 2) + kv];
      unsigned short h0, h1, h2, h3, l0, l1, l2, l3;
      split2(f.x, h0, l0); split2(f.y, h1, l1);
      split2(f.z, h2, l2); split2(f.w, h3, l3);
      uint2 ph; ph.x = (unsigned)h0 | ((unsigned)h1 << 16);
                ph.y = (unsigned)h2 | ((unsigned)h3 << 16);
      uint2 pl; pl.x = (unsigned)l0 | ((unsigned)l1 << 16);
                pl.y = (unsigned)l2 | ((unsigned)l3 << 16);
      *(uint2*)(Ahi + (size_t)r * LDA + kv * 4) = ph;
      *(uint2*)(Alo + (size_t)r * LDA + kv * 4) = pl;
    }
  }
  __syncthreads();

  // --- wave tiling: 4 row strips x 2 column tiles ---
  const int strip = wave & 3;            // 16-row strip
  const int ctile = wave >> 2;           // 16-col tile within the 32-col panel
  const int rbase = strip * 16;
  const int mycol = ctile * 16 + m;

  // row labels for my 8 accumulator rows (C layout: row = v + 8*half)
  int labR[8];
#pragma unroll
  for (int v = 0; v < 8; ++v)
    labR[v] = (int)labels[row0 + rbase + v + 8 * half];

  float negTot[8];
#pragma unroll
  for (int v = 0; v < 8; ++v) negTot[v] = 0.f;

  const unsigned short* Abase_h = Ahi + (size_t)(rbase + m) * LDA;
  const unsigned short* Abase_l = Alo + (size_t)(rbase + m) * LDA;

  for (int pass = 0; pass < 2; ++pass) {
    float accRow[8];
#pragma unroll
    for (int v = 0; v < 8; ++v) accRow[v] = 0.f;

    // prologue: stage panel 0 into buffer 0
    stage_panel(feats, 0, Bbuf, Bbuf + (size_t)PAN * LDA, tid);
    __syncthreads();

    for (int pan = 0; pan < NPAN; ++pan) {
      const int cur = pan & 1;
      unsigned short* curH = Bbuf + (size_t)(2 * cur) * PAN * LDA;
      unsigned short* curL = curH + (size_t)PAN * LDA;
      if (pan + 1 < NPAN) {
        unsigned short* nxtH = Bbuf + (size_t)(2 * (cur ^ 1)) * PAN * LDA;
        stage_panel(feats, pan + 1, nxtH, nxtH + (size_t)PAN * LDA, tid);
      }

      // ---- 16x16 tile GEMM over K=512, split-bf16 (3 WMMAs per k-step) ----
      // Three INDEPENDENT accumulator chains (hi*hi, hi*lo, lo*hi) so the
      // matrix pipe sees 3-way ILP instead of one 48-deep RAW chain.
      v8f accA, accB, accC;
#pragma unroll
      for (int v = 0; v < 8; ++v) { accA[v] = 0.f; accB[v] = 0.f; accC[v] = 0.f; }

      const unsigned short* Bcol_h = curH + (size_t)mycol * LDA;
      const unsigned short* Bcol_l = curL + (size_t)mycol * LDA;
#pragma unroll
      for (int ks = 0; ks < NKS; ++ks) {
        const int kk = ks * 32;
        // A layout (16-bit 16x32): lane m holds row m; K runs
        //   [kk+half*8, +8) and [kk+half*8+16, +8)
        const unsigned short* pa = Abase_h + kk + half * 8;
        const unsigned short* pl = Abase_l + kk + half * 8;
        v16bf a_h = ldmat(pa, pa + 16);
        v16bf a_l = ldmat(pl, pl + 16);
        // B layout (16-bit 32x16): lane holds col m, K = [kk+half*8, +8) and [kk+half*8+16, +8)
        const unsigned short* pb  = Bcol_h + kk + half * 8;
        const unsigned short* pbl = Bcol_l + kk + half * 8;
        v16bf b_h = ldmat(pb, pb + 16);
        v16bf b_l = ldmat(pbl, pbl + 16);
        accA = wmma_bf16(a_h, b_h, accA);   // hi*hi chain
        accB = wmma_bf16(a_h, b_l, accB);   // hi*lo chain
        accC = wmma_bf16(a_l, b_h, accC);   // lo*hi chain
      }

      // ---- register epilogue (no LDS): accumulate per-row partials ----
      const int colg = pan * PAN + mycol;
      const int labC = (int)labels[colg];
#pragma unroll
      for (int v = 0; v < 8; ++v) {
        const float sim = (accA[v] + accB[v]) + accC[v];
        const float e   = __expf(sim * INV_T);
        const bool  eq  = (labR[v] == labC);
        if (pass == 0) {
          accRow[v] += eq ? 0.f : e;                       // neg_exp partial
        } else {
          const int rowg = row0 + rbase + v + 8 * half;
          if (eq && rowg != colg) {
            const float pt = e / (e + negTot[v] + EPSV);
            const float om = 1.f - pt;
            accRow[v] += -(om * om) * __logf(pt + EPSV);   // focal term, gamma=2
          }
        }
      }
      __syncthreads();   // next buffer staged; this buffer safe to restage
    }

    // reduce 16 lanes of each half (xor stays inside the half for masks<16)
#pragma unroll
    for (int v = 0; v < 8; ++v) {
      accRow[v] += __shfl_xor(accRow[v], 1, 32);
      accRow[v] += __shfl_xor(accRow[v], 2, 32);
      accRow[v] += __shfl_xor(accRow[v], 4, 32);
      accRow[v] += __shfl_xor(accRow[v], 8, 32);
    }
    if (m == 0) {
#pragma unroll
      for (int v = 0; v < 8; ++v)
        rparts[ctile * ROWS + rbase + v + 8 * half] = accRow[v];
    }
    __syncthreads();
    if (pass == 0) {
      // combine the two column-tile waves in FIXED order (deterministic)
#pragma unroll
      for (int v = 0; v < 8; ++v) {
        const int rl = rbase + v + 8 * half;
        negTot[v] = rparts[rl] + rparts[ROWS + rl];
      }
      __syncthreads();
    }
  }

  // --- per-row loss, fixed-order block reduction ---
  if (tid < ROWS) {
    const float rs  = rparts[tid] + rparts[ROWS + tid];
    const int   lab = (int)labels[row0 + tid];
    const int   cnt = hist[lab];
    const float pos = (float)(cnt - 1);
    const float neg = (float)(BB - cnt);
    lossbuf[tid] = (pos > 0.f && neg > 0.f) ? rs / fmaxf(pos, 1.f) : 0.f;
  }
  __syncthreads();
  if (tid == 0) {
    float s = 0.f;
    for (int i = 0; i < ROWS; ++i) s += lossbuf[i];
    partials[blk] = s;
  }
}

// Fixed-order final sum over 128 block partials -> mean loss.
__global__ void fcl_finalize(const float* __restrict__ partials,
                             float* __restrict__ out, int n) {
  if (blockIdx.x == 0 && threadIdx.x == 0) {
    float s = 0.f;
    for (int i = 0; i < n; ++i) s += partials[i];
    out[0] = s / (float)BB;
  }
}

// ---------------------------------------------------------------------------
extern "C" void kernel_launch(void* const* d_in, const int* in_sizes, int n_in,
                              void* d_out, int out_size, void* d_ws, size_t ws_size,
                              hipStream_t stream) {
  const float*     feats  = (const float*)d_in[0];       // [8192, 512] f32
  const long long* labels = (const long long*)d_in[1];   // [8192] int64
  float* out      = (float*)d_out;                       // scalar f32
  float* partials = (float*)d_ws;                        // 128 floats of scratch

  // LDS: A hi/lo (64x520 x2) + B double-buffer hi/lo (2 x 2 x 32x520)
  //      + rparts(128f) + lossbuf(64f) + hist(16i)  ~= 267 KB  (< 320 KB/WGP)
  const size_t smem =
      (size_t)2 * ROWS * LDA * sizeof(unsigned short) +
      (size_t)4 * PAN  * LDA * sizeof(unsigned short) +
      (2 * ROWS + ROWS) * sizeof(float) + 16 * sizeof(int);

  (void)hipFuncSetAttribute(reinterpret_cast<const void*>(fcl_main),
                            hipFuncAttributeMaxDynamicSharedMemorySize,
                            (int)smem);

  fcl_main<<<BB / ROWS, 256, smem, stream>>>(feats, labels, partials);
  fcl_finalize<<<1, 32, 0, stream>>>(partials, out, BB / ROWS);
}
